// ESMoE_10909216932614
// MI455X (gfx1250) — matrix operand order repaired and checked by reading, not compiled, not run on hardware
//
#include <hip/hip_runtime.h>
#include <hip/hip_bf16.h>

// ---------------------------------------------------------------------------
// Types for CDNA5 WMMA / TDM
// ---------------------------------------------------------------------------
typedef __attribute__((ext_vector_type(16))) __bf16        v16bf;
typedef __attribute__((ext_vector_type(8)))  float         v8f;
typedef __attribute__((ext_vector_type(8)))  unsigned int  v8u;
typedef __attribute__((ext_vector_type(4)))  unsigned int  v4u;
typedef __attribute__((ext_vector_type(8)))  int           v8i;
typedef __attribute__((ext_vector_type(4)))  int           v4i;

#define WMMA_BF16(a, b, c) \
    __builtin_amdgcn_wmma_f32_16x16x32_bf16(false, (a), false, (b), (short)0, (c), false, false)

// Problem constants
#define BATCH   16
#define CIN     256
#define HH      56
#define WW      56
#define PIX     3136      // 56*56
#define NEXP    4
#define HID     512
#define OUT_C   256
#define RED     32
#define PH      14
#define PW      14

// Main-kernel tiling
#define NT      32                 // pixels per block tile
#define NTILES  (PIX / NT)         // 98
#define RSX     (CIN + 8)          // LDS row stride (bf16 elems), X tile
#define RSH     (HID + 8)          // LDS row stride (bf16 elems), H tile
#define TP      64                 // transpose tile

// ---------------------------------------------------------------------------
// Helpers
// ---------------------------------------------------------------------------
__device__ __forceinline__ unsigned short f32_to_bf16(float f) {
    unsigned int u = __builtin_bit_cast(unsigned int, f);
    u = (u + 0x7FFFu + ((u >> 16) & 1u)) >> 16;   // round-to-nearest-even
    return (unsigned short)u;
}

__device__ __forceinline__ unsigned int bf16pk(float a, float b) {
    return (unsigned int)f32_to_bf16(a) | ((unsigned int)f32_to_bf16(b) << 16);
}

__device__ __forceinline__ float silu_f(float v) {
    // v * sigmoid(v), fast reciprocal (v_rcp_f32) instead of IEEE divide
    return v * __builtin_amdgcn_rcpf(1.0f + __expf(-v));
}

// Packed-A fragment load: weights were pre-packed so each (16m x 32k) fragment
// is 32 lanes x 16 bf16 (32B) contiguous -> two coalesced b128 loads, no VALU.
__device__ __forceinline__ v16bf load_a_pk(const unsigned int* __restrict__ Wp,
                                           int kb /* K/32 */, int m0, int k0, int lane) {
    size_t fb = (size_t)(m0 >> 4) * kb + (k0 >> 5);
    const uint4* p = (const uint4*)(Wp + (fb * 32 + lane) * 8);
    uint4 lo = p[0], hi4 = p[1];
    v8u r;
    r[0] = lo.x;  r[1] = lo.y;  r[2] = lo.z;  r[3] = lo.w;
    r[4] = hi4.x; r[5] = hi4.y; r[6] = hi4.z; r[7] = hi4.w;
    return __builtin_bit_cast(v16bf, r);
}

// B-fragment (32x16 bf16) from pixel-major bf16 LDS tile: 16 contiguous bf16
// per lane -> two ds_load_b128.
__device__ __forceinline__ v16bf load_b_frag(const unsigned short* lds,
                                             int rs, int n0, int k0, int lane) {
    int ln = lane & 15;
    int hi = lane >> 4;
    const uint4* q = (const uint4*)(lds + (size_t)(n0 + ln) * rs + k0 + 16 * hi);
    uint4 a = q[0], b = q[1];
    v8u r;
    r[0] = a.x; r[1] = a.y; r[2] = a.z; r[3] = a.w;
    r[4] = b.x; r[5] = b.y; r[6] = b.z; r[7] = b.w;
    return __builtin_bit_cast(v16bf, r);
}

// ---------------------------------------------------------------------------
// One-time weight pre-pack: fp32 [M,K] row-major -> bf16 A-fragment order.
// Fragment fb = (m0/16)*(K/32) + k0/32; within fragment, lane-major 16 bf16
// contiguous K order (lanes 0-15: M=lane, K=0..15; 16-31: K=16..31).
// ---------------------------------------------------------------------------
__global__ __launch_bounds__(256) void pack_weights(const float* __restrict__ W,
                                                    unsigned int* __restrict__ Wp,
                                                    int M, int K) {
    int idx = blockIdx.x * 256 + threadIdx.x;
    int nth = (M >> 4) * (K >> 5) * 32;
    if (idx >= nth) return;
    int fb   = idx >> 5;
    int lane = idx & 31;
    int kb   = K >> 5;
    int m0   = (fb / kb) << 4;
    int k0   = (fb % kb) << 5;
    int lm = lane & 15, hi = lane >> 4;
    const float* rp = W + (size_t)(m0 + lm) * K;
    int b1 = k0 + 16 * hi;
    int b2 = k0 + 16 * hi + 8;
    unsigned int* dst = Wp + (size_t)idx * 8;
#pragma unroll
    for (int t = 0; t < 4; ++t) dst[t]     = bf16pk(rp[b1 + 2 * t], rp[b1 + 2 * t + 1]);
#pragma unroll
    for (int t = 0; t < 4; ++t) dst[4 + t] = bf16pk(rp[b2 + 2 * t], rp[b2 + 2 * t + 1]);
}

// ---------------------------------------------------------------------------
// One-time activation transform: x[B,C,56,56] fp32 -> xt[B,P,C] bf16
// (pixel-major so every 32-pixel tile is one contiguous 16KB block for TDM).
// LDS tile transpose keeps both global streams coalesced.
// ---------------------------------------------------------------------------
__global__ __launch_bounds__(256) void transpose_x(const float* __restrict__ x,
                                                   unsigned short* __restrict__ xt) {
    __shared__ unsigned short Tl[TP * (TP + 2)];
    int b  = blockIdx.z;
    int c0 = blockIdx.y * TP;
    int p0 = blockIdx.x * TP;
    int tid = threadIdx.x;
#pragma unroll
    for (int t = 0; t < (TP * TP) / 256; ++t) {
        int idx = t * 256 + tid;
        int i = idx >> 6;         // channel in tile
        int j = idx & 63;         // pixel in tile (coalesced reads)
        float v = x[((size_t)b * CIN + c0 + i) * PIX + p0 + j];
        Tl[i * (TP + 2) + j] = f32_to_bf16(v);
    }
    __syncthreads();
#pragma unroll
    for (int t = 0; t < (TP * TP) / 256; ++t) {
        int idx = t * 256 + tid;
        int p  = idx >> 6;        // pixel in tile
        int cc = idx & 63;        // channel in tile (coalesced writes)
        xt[((size_t)b * PIX + p0 + p) * CIN + c0 + cc] = Tl[cc * (TP + 2) + p];
    }
}

// ---------------------------------------------------------------------------
// Router stage 1: 4x4 average pool  x[B,C,56,56] -> xp[B,C,14,14]
// ---------------------------------------------------------------------------
__global__ __launch_bounds__(256) void pool_kernel(const float* __restrict__ x,
                                                   float* __restrict__ xp) {
    int idx = blockIdx.x * 256 + threadIdx.x;
    const int total = BATCH * CIN * PH * PW;
    if (idx >= total) return;
    int j = idx % PW;
    int i = (idx / PW) % PH;
    int c = (idx / (PH * PW)) % CIN;
    int b = idx / (PH * PW * CIN);
    const float* px = x + (((size_t)b * CIN + c) * HH + i * 4) * WW + j * 4;
    float s = 0.0f;
#pragma unroll
    for (int dy = 0; dy < 4; ++dy)
#pragma unroll
        for (int dx = 0; dx < 4; ++dx) s += px[dy * WW + dx];
    xp[idx] = s * (1.0f / 16.0f);
}

// ---------------------------------------------------------------------------
// Router stage 2: 3x3 conv(red=32) + SiLU + 1x1->E + spatial mean + softmax
//                 + top-2 renorm -> dense weights wdense[B,E]
// ---------------------------------------------------------------------------
__global__ __launch_bounds__(256) void router_kernel(const float* __restrict__ xp,
                                                     const float* __restrict__ rw1,
                                                     const float* __restrict__ rw2,
                                                     float* __restrict__ wdense) {
    __shared__ float sred[NEXP * 256];
    int b   = blockIdx.x;
    int tid = threadIdx.x;
    float partial[NEXP] = {0.f, 0.f, 0.f, 0.f};

    const float* xpb = xp + (size_t)b * CIN * PH * PW;

    for (int idx = tid; idx < RED * PH * PW; idx += 256) {
        int r  = idx / (PH * PW);
        int pe = idx % (PH * PW);
        int i  = pe / PW;
        int j  = pe % PW;
        float acc = 0.0f;
        for (int di = 0; di < 3; ++di) {
            int ii = i + di - 1;
            if (ii < 0 || ii >= PH) continue;
            for (int dj = 0; dj < 3; ++dj) {
                int jj = j + dj - 1;
                if (jj < 0 || jj >= PW) continue;
                const float* xc = xpb + ii * PW + jj;
                const float* wc = rw1 + ((size_t)r * CIN * 9) + di * 3 + dj;
                for (int c = 0; c < CIN; ++c)
                    acc += xc[c * (PH * PW)] * wc[(size_t)c * 9];
            }
        }
        float h = silu_f(acc);
#pragma unroll
        for (int e = 0; e < NEXP; ++e) partial[e] += rw2[e * RED + r] * h;
    }

#pragma unroll
    for (int e = 0; e < NEXP; ++e) sred[e * 256 + tid] = partial[e];
    __syncthreads();
    for (int off = 128; off > 0; off >>= 1) {
        if (tid < off) {
#pragma unroll
            for (int e = 0; e < NEXP; ++e)
                sred[e * 256 + tid] += sred[e * 256 + tid + off];
        }
        __syncthreads();
    }

    if (tid == 0) {
        float lg[NEXP];
#pragma unroll
        for (int e = 0; e < NEXP; ++e) lg[e] = sred[e * 256] * (1.0f / (PH * PW));
        float m = lg[0];
#pragma unroll
        for (int e = 1; e < NEXP; ++e) m = fmaxf(m, lg[e]);
        float ex[NEXP], s = 0.0f;
#pragma unroll
        for (int e = 0; e < NEXP; ++e) { ex[e] = __expf(lg[e] - m); s += ex[e]; }
        float inv = __builtin_amdgcn_rcpf(s);
        float pr[NEXP];
#pragma unroll
        for (int e = 0; e < NEXP; ++e) pr[e] = ex[e] * inv;
        int i1 = 0;
#pragma unroll
        for (int e = 1; e < NEXP; ++e) if (pr[e] > pr[i1]) i1 = e;
        int i2 = (i1 == 0) ? 1 : 0;
#pragma unroll
        for (int e = 0; e < NEXP; ++e)
            if (e != i1 && pr[e] > pr[i2]) i2 = e;
        float t = pr[i1] + pr[i2] + 1e-6f;
#pragma unroll
        for (int e = 0; e < NEXP; ++e) wdense[b * NEXP + e] = 0.0f;
        wdense[b * NEXP + i1] = pr[i1] / t;
        wdense[b * NEXP + i2] = pr[i2] / t;
    }
}

// ---------------------------------------------------------------------------
// Main fused MoE kernel. Per (batch, 32-pixel tile) block, 8 wave32.
// X tile arrives via Tensor Data Mover (tensor_load_to_lds) with hardware
// row padding; all GEMMs via v_wmma_f32_16x16x32_bf16 on pre-packed weights.
// ---------------------------------------------------------------------------
__global__ __launch_bounds__(256) void moe_main(const unsigned short* __restrict__ xt,
                                                const unsigned int* __restrict__ w1p,
                                                const unsigned int* __restrict__ w2p,
                                                const unsigned int* __restrict__ swp,
                                                const float* __restrict__ wdense,
                                                float* __restrict__ out) {
    __shared__ unsigned short Xl[NT * RSX];   // 32 x 264 bf16 = 16.9 KB (TDM-padded)
    __shared__ unsigned short Hl[NT * RSH];   // 32 x 520 bf16 = 33.3 KB

    int tid  = threadIdx.x;
    int lane = tid & 31;
    int wv   = tid >> 5;
    int b    = blockIdx.x / NTILES;
    int p0   = (blockIdx.x % NTILES) * NT;

    // ---- TDM: DMA the contiguous 32x256 bf16 X tile into LDS --------------
    // D# per CDNA5 ISA; pad_enable inserts 16B after
    // every 512B row -> LDS row stride RSX = 264 bf16.
    if (wv == 0) {
        const unsigned short* gX = xt + ((size_t)b * PIX + p0) * CIN;
        unsigned long long ga = (unsigned long long)(size_t)gX;
        unsigned ldsa = (unsigned)(size_t)(&Xl[0]);   // low 32 bits = LDS byte addr
        v4u g0;
        g0[0] = 1u;                                   // count=1, user descriptor
        g0[1] = ldsa;                                 // lds_addr
        g0[2] = (unsigned)(ga & 0xFFFFFFFFu);         // global_addr[31:0]
        g0[3] = (unsigned)((ga >> 32) & 0x01FFFFFFu)  // global_addr[56:32]
              | (2u << 30);                           // type = 2 ("image")
        v8i g1;
        g1[0] = (int)((1u << 16)      // data_size = 2 bytes
                    | (1u << 20)      // pad_enable
                    | (6u << 22)      // pad_interval: 128 dwords (512B row)
                    | (3u << 25));    // pad_amount: 4 dwords (16B)
        g1[1] = (int)(((unsigned)CIN & 0xFFFFu) << 16);   // tensor_dim0 lo16
        g1[2] = (int)(((unsigned)PIX & 0xFFFFu) << 16);   // dim0 hi=0 | tensor_dim1 lo16
        g1[3] = (int)(((unsigned)CIN) << 16);             // dim1 hi=0 | tile_dim0 = 256
        g1[4] = (int)NT;                                  // tile_dim1 = 32, tile_dim2 = 0
        g1[5] = (int)CIN;                                 // tensor_dim0_stride = 256
        g1[6] = 0;
        g1[7] = 0;
        v4i gz4 = {};
        v8i gz8 = {};
        __builtin_amdgcn_tensor_load_to_lds(g0, g1, gz4, gz4, gz8, 0);
        __builtin_amdgcn_s_wait_tensorcnt(0);
    }
    __syncthreads();

    // ---- shared expert GEMM: rows [32*wv, 32*wv+32), cols [0,32) ----------
    v8f accO[2][2];
    v8f zeroF = {};
#pragma unroll
    for (int i = 0; i < 2; ++i)
#pragma unroll
        for (int j = 0; j < 2; ++j) accO[i][j] = zeroF;

#pragma unroll
    for (int k0 = 0; k0 < CIN; k0 += 32) {
        v16bf b0 = load_b_frag(Xl, RSX, 0,  k0, lane);
        v16bf b1 = load_b_frag(Xl, RSX, 16, k0, lane);
        v16bf a0 = load_a_pk(swp, CIN / 32, wv * 32,      k0, lane);
        v16bf a1 = load_a_pk(swp, CIN / 32, wv * 32 + 16, k0, lane);
        accO[0][0] = WMMA_BF16(a0, b0, accO[0][0]);
        accO[0][1] = WMMA_BF16(a0, b1, accO[0][1]);
        accO[1][0] = WMMA_BF16(a1, b0, accO[1][0]);
        accO[1][1] = WMMA_BF16(a1, b1, accO[1][1]);
    }
#pragma unroll
    for (int i = 0; i < 2; ++i)
#pragma unroll
        for (int j = 0; j < 2; ++j)
#pragma unroll
            for (int t = 0; t < 8; ++t) accO[i][j][t] = silu_f(accO[i][j][t]);

    // ---- expert loop (uniform branch per block) ---------------------------
    const float* wdb = wdense + b * NEXP;
    int ln = lane & 15;
    int hi = lane >> 4;

    for (int e = 0; e < NEXP; ++e) {
        float we = wdb[e];
        if (we == 0.0f) continue;   // uniform across block

        // H = silu(W1_e @ X): rows [64*wv, 64*wv+64); w1p packed as M=2048,K=256
        v8f accH[4][2];
#pragma unroll
        for (int i = 0; i < 4; ++i)
#pragma unroll
            for (int j = 0; j < 2; ++j) accH[i][j] = zeroF;

#pragma unroll
        for (int k0 = 0; k0 < CIN; k0 += 32) {
            v16bf b0 = load_b_frag(Xl, RSX, 0,  k0, lane);
            v16bf b1 = load_b_frag(Xl, RSX, 16, k0, lane);
#pragma unroll
            for (int i = 0; i < 4; ++i) {
                v16bf a = load_a_pk(w1p, CIN / 32, e * HID + wv * 64 + 16 * i, k0, lane);
                accH[i][0] = WMMA_BF16(a, b0, accH[i][0]);
                accH[i][1] = WMMA_BF16(a, b1, accH[i][1]);
            }
        }

        __syncthreads();   // previous expert's W2 GEMM done reading Hl
#pragma unroll
        for (int i = 0; i < 4; ++i)
#pragma unroll
            for (int j = 0; j < 2; ++j) {
                v8f h = accH[i][j];
#pragma unroll
                for (int t = 0; t < 8; ++t) h[t] = silu_f(h[t]);
                int prow = 16 * j + ln;
                int mst  = wv * 64 + 16 * i + 8 * hi;
                uint4 pk;
                pk.x = bf16pk(h[0], h[1]);
                pk.y = bf16pk(h[2], h[3]);
                pk.z = bf16pk(h[4], h[5]);
                pk.w = bf16pk(h[6], h[7]);
                *(uint4*)(&Hl[prow * RSH + mst]) = pk;
            }
        __syncthreads();

        // Y = W2_e @ H: rows [32*wv, 32*wv+32), K=512; w2p packed M=1024,K=512
        v8f accY[2][2];
#pragma unroll
        for (int i = 0; i < 2; ++i)
#pragma unroll
            for (int j = 0; j < 2; ++j) accY[i][j] = zeroF;

#pragma unroll 4
        for (int k0 = 0; k0 < HID; k0 += 32) {
            v16bf b0 = load_b_frag(Hl, RSH, 0,  k0, lane);
            v16bf b1 = load_b_frag(Hl, RSH, 16, k0, lane);
            v16bf a0 = load_a_pk(w2p, HID / 32, e * OUT_C + wv * 32,      k0, lane);
            v16bf a1 = load_a_pk(w2p, HID / 32, e * OUT_C + wv * 32 + 16, k0, lane);
            accY[0][0] = WMMA_BF16(a0, b0, accY[0][0]);
            accY[0][1] = WMMA_BF16(a0, b1, accY[0][1]);
            accY[1][0] = WMMA_BF16(a1, b0, accY[1][0]);
            accY[1][1] = WMMA_BF16(a1, b1, accY[1][1]);
        }
#pragma unroll
        for (int i = 0; i < 2; ++i)
#pragma unroll
            for (int j = 0; j < 2; ++j) accO[i][j] += we * accY[i][j];
    }

    // ---- store output -----------------------------------------------------
    float* ob = out + ((size_t)b * OUT_C) * PIX + p0;
#pragma unroll
    for (int i = 0; i < 2; ++i)
#pragma unroll
        for (int j = 0; j < 2; ++j) {
            int m = wv * 32 + 16 * i + 8 * hi;
            int p = 16 * j + ln;
#pragma unroll
            for (int t = 0; t < 8; ++t)
                ob[(size_t)(m + t) * PIX + p] = accO[i][j][t];
        }
}

// ---------------------------------------------------------------------------
// Launch
// ---------------------------------------------------------------------------
extern "C" void kernel_launch(void* const* d_in, const int* in_sizes, int n_in,
                              void* d_out, int out_size, void* d_ws, size_t ws_size,
                              hipStream_t stream) {
    const float* x   = (const float*)d_in[0];
    const float* rw1 = (const float*)d_in[1];
    const float* rw2 = (const float*)d_in[2];
    const float* ew1 = (const float*)d_in[3];
    const float* ew2 = (const float*)d_in[4];
    const float* sw  = (const float*)d_in[5];
    float*       out = (float*)d_out;

    // Workspace layout (bytes)
    char* ws = (char*)d_ws;
    float*          wdense = (float*)ws;                               // 64 f32
    float*          xp     = (float*)(ws + 256);                       // 802816 f32
    unsigned short* xt     = (unsigned short*)(ws + 256 + 3211264);    // 25.7 MB
    unsigned int*   swp    = (unsigned int*)(ws + 256 + 3211264 + 25690112);
    unsigned int*   w1p    = (unsigned int*)((char*)swp + 131072);     // 1 MB
    unsigned int*   w2p    = (unsigned int*)((char*)w1p + 1048576);    // 1 MB

    // One-time transforms (graph-captured, so they replay each call; they are
    // pure functions of the inputs -> deterministic).
    pack_weights<<<(256/16)*(256/32)*32/256,        256, 0, stream>>>(sw,  swp, OUT_C,        CIN);
    pack_weights<<<(2048/16)*(256/32)*32/256,       256, 0, stream>>>(ew1, w1p, NEXP * HID,   CIN);
    pack_weights<<<(1024/16)*(512/32)*32/256,       256, 0, stream>>>(ew2, w2p, NEXP * OUT_C, HID);
    transpose_x <<<dim3(PIX / TP, CIN / TP, BATCH), 256, 0, stream>>>(x, xt);

    pool_kernel  <<<(BATCH * CIN * PH * PW + 255) / 256, 256, 0, stream>>>(x, xp);
    router_kernel<<<BATCH, 256, 0, stream>>>(xp, rw1, rw2, wdense);

    moe_main<<<BATCH * NTILES, 256, 0, stream>>>(xt, w1p, w2p, swp, wdense, out);
}
